// MDTA_25228637897180
// MI455X (gfx1250) — matrix operand order repaired and checked by reading, not compiled, not run on hardware
//
#include <hip/hip_runtime.h>
#include <math.h>
#include <stdint.h>

// ---------------- CDNA5 WMMA / TDM types ----------------
typedef __bf16 v16bf __attribute__((ext_vector_type(16)));
typedef float  v8f   __attribute__((ext_vector_type(8)));
typedef unsigned int v4u __attribute__((ext_vector_type(4)));
typedef int v8i __attribute__((ext_vector_type(8)));
typedef int v4i __attribute__((ext_vector_type(4)));

#define DIM    64
#define DHEAD  16
#define HEADS  4
#define INNER  64
#define BB     4
#define HH     256
#define WW     256
#define HW     65536
#define QKVC   192   // 3 * INNER

// ---------------- fragment helpers (contiguous k = 16*half + i, same for A and B) -------
// A matrix 16x32 bf16, row-major LDS [M][K-ld]:
//   lanes 0-15: M=lane;  elem i -> K=i
//   lanes16-31: M=lane-16; elem i -> K=16+i
__device__ inline v16bf frag_a_16x32(const __bf16* lds, int ld, int row0, int col0, int lane) {
    int half = (lane >> 4) & 1;
    int m    = lane & 15;
    const __bf16* r = lds + (size_t)(row0 + m) * ld + col0;
    v16bf a;
#pragma unroll
    for (int i = 0; i < 16; ++i) a[i] = r[half * 16 + i];           // K = 16*half + i
    return a;
}

// B matrix 32x16 bf16 where LDS is row-major [K][N-ld]:
//   lanes 0-15: N=lane, elem i -> K=i ; lanes 16-31: N=lane-16, elem i -> K=16+i
__device__ inline v16bf frag_b_rowK(const __bf16* lds, int ld, int k0, int n0, int lane) {
    int half = (lane >> 4) & 1;
    int n    = lane & 15;
    v16bf b;
#pragma unroll
    for (int i = 0; i < 16; ++i)
        b[i] = lds[(size_t)(k0 + half * 16 + i) * ld + (n0 + n)];
    return b;
}

// B matrix 32x16 bf16 where LDS is row-major [N][K-ld]  (i.e. B = k^T for Gram)
__device__ inline v16bf frag_b_colK(const __bf16* lds, int ld, int k0, int n0, int lane) {
    int half = (lane >> 4) & 1;
    int n    = lane & 15;
    const __bf16* r = lds + (size_t)(n0 + n) * ld + k0 + half * 16;
    v16bf b;
#pragma unroll
    for (int i = 0; i < 16; ++i) b[i] = r[i];
    return b;
}

__device__ inline v8f wmma_bf16(v16bf a, v16bf b, v8f c) {
    return __builtin_amdgcn_wmma_f32_16x16x32_bf16(false, a, false, b, (short)0, c, false, false);
}

// ---------------- Tensor Data Mover: 2-D tile load, bf16 elements -----------
// D# per CDNA5 ISA. Groups 2/3 zero (tile_dim2==0 -> 2D).
// LDS row padding via pad_enable: interval code k -> 2^(k+1) DWORDs, amount
// code a -> (a+1) DWORDs appended to LDS address.
#if __has_builtin(__builtin_amdgcn_tensor_load_to_lds)
#define HAVE_TDM 1
__device__ inline void tdm_load_2d_bf16(uint32_t lds_addr, const void* gptr,
                                        uint32_t tensor_d0, uint32_t tensor_d1,
                                        uint32_t tile_d0, uint32_t tile_d1,
                                        uint64_t stride0_elems,
                                        uint32_t pad_interval_code,
                                        uint32_t pad_amount_code) {
    uint64_t ga = (uint64_t)(uintptr_t)gptr;
    v4u g0;
    g0[0] = 1u;                                            // count=1, user D#
    g0[1] = lds_addr;                                      // LDS byte address
    g0[2] = (uint32_t)(ga & 0xffffffffu);                  // global_addr[31:0]
    g0[3] = (uint32_t)((ga >> 32) & 0x01ffffffu) | (2u << 30); // [56:32] | type=2
    v8i g1;
    g1[0] = (int)((1u << 16)                               // data_size = 2 bytes
                | (1u << 20)                               // pad_enable
                | (pad_interval_code << 22)
                | (pad_amount_code << 25));
    g1[1] = (int)((tensor_d0 & 0xffffu) << 16);            // tensor_dim0[15:0]
    g1[2] = (int)(((tensor_d0 >> 16) & 0xffffu)            // tensor_dim0[31:16]
                | ((tensor_d1 & 0xffffu) << 16));          // tensor_dim1[15:0]
    g1[3] = (int)(((tensor_d1 >> 16) & 0xffffu)            // tensor_dim1[31:16]
                | ((tile_d0 & 0xffffu) << 16));            // tile_dim0
    g1[4] = (int)(tile_d1 & 0xffffu);                      // tile_dim1 (tile_dim2=0)
    g1[5] = (int)(uint32_t)(stride0_elems & 0xffffffffu);  // dim0_stride[31:0]
    g1[6] = (int)(uint32_t)((stride0_elems >> 32) & 0xffffu);
    g1[7] = 0;
    v4i gz = {0, 0, 0, 0};
#if defined(__clang_major__) && (__clang_major__ >= 23)
    v8i gz8 = {0, 0, 0, 0, 0, 0, 0, 0};
    __builtin_amdgcn_tensor_load_to_lds(g0, g1, gz, gz, gz8, 0);
#else
    __builtin_amdgcn_tensor_load_to_lds(g0, g1, gz, gz, 0);
#endif
}
__device__ inline void tdm_wait0() {
#if __has_builtin(__builtin_amdgcn_s_wait_tensorcnt)
    __builtin_amdgcn_s_wait_tensorcnt(0);
#else
    asm volatile("s_wait_tensorcnt 0x0" ::: "memory");
#endif
}
#else
#define HAVE_TDM 0
#endif

// =====================================================================
// K1: qkv = Wqkv(192x64) @ x(64 x HW) per batch.  bf16 WMMA, f32 acc.
// grid (HW/128, B), block 256 (8 waves). 128-pixel tile per block.
// =====================================================================
__global__ __launch_bounds__(256) void k1_qkv_gemm(const float* __restrict__ x_in,
                                                   const float* __restrict__ w_qkv,
                                                   __bf16* __restrict__ qkv_bf) {
    __shared__ __bf16 lwq[QKVC][72];     // A: [192][64] (pad 72)
    __shared__ __bf16 lx[DIM][136];      // B: [64][128] (pad 136)

    const int tid  = threadIdx.x;
    const int lane = tid & 31;
    const int wv   = tid >> 5;
    const int b    = blockIdx.y;
    const int p0   = blockIdx.x * 128;

    // stage A = Wqkv (f32 -> bf16)
    for (int i = tid; i < QKVC * DIM; i += 256)
        lwq[i >> 6][i & 63] = (__bf16)w_qkv[i];
    // stage B = x tile (f32 -> bf16)
    for (int i = tid; i < DIM * 128; i += 256) {
        int c = i >> 7, p = i & 127;
        lx[c][p] = (__bf16)x_in[((size_t)(b * DIM + c) << 16) + p0 + p];
    }
    __syncthreads();

    const int nn = wv;                       // N-subtile (16 pixels) per wave
    for (int m = 0; m < 12; ++m) {           // 12 M-tiles of 16 rows
        v8f acc = {};
#pragma unroll
        for (int k0 = 0; k0 < 64; k0 += 32) {
            v16bf a = frag_a_16x32(&lwq[0][0], 72, m * 16, k0, lane);
            v16bf bb = frag_b_rowK(&lx[0][0], 136, k0, nn * 16, lane);
            acc = wmma_bf16(a, bb, acc);
        }
        // D layout: VGPR r -> M = r + 8*half ; N = lane&15
        int half = (lane >> 4) & 1;
        int pix  = p0 + nn * 16 + (lane & 15);
#pragma unroll
        for (int r = 0; r < 8; ++r) {
            int o = m * 16 + r + 8 * half;
            qkv_bf[((size_t)(b * QKVC + o) << 16) + pix] = (__bf16)acc[r];
        }
    }
}

// =====================================================================
// K2: depthwise 3x3 on qkv -> route:
//      q,k  -> LDS tiles -> WMMA Gram G[b,n,16,16] (atomic f32 reduce)
//              + per-channel sum(x^2) for L2 norms
//      v    -> v * illu  -> bf16 scratch vi[b,64,HW]
// grid (W/128, H, B), block 128 (4 waves), one 128-pixel half-row per block
// (halo origin can be out-of-bounds low -> bounds-checked manual staging,
//  TDM OOB only handles the upper bound)
// =====================================================================
__global__ __launch_bounds__(128) void k2_dw_gram(const __bf16* __restrict__ qkv_bf,
                                                  const float* __restrict__ w_dw,
                                                  const float* __restrict__ illu,
                                                  __bf16* __restrict__ vi_bf,
                                                  float* __restrict__ G,
                                                  float* __restrict__ sumsq) {
    __shared__ __bf16 lq[INNER][136];    // q values [ch][pix]
    __shared__ __bf16 lk[INNER][136];    // k values [ch][pix]
    __shared__ __bf16 lh[8][3][132];     // halo stage: 8 ch x 3 rows x 130 cols
    __shared__ float  lwd[8][9];

    const int tid  = threadIdx.x;
    const int lane = tid & 31;
    const int wv   = tid >> 5;
    const int b    = blockIdx.z;
    const int y    = blockIdx.y;
    const int x0   = blockIdx.x * 128;
    const int x    = x0 + tid;

    for (int g = 0; g < QKVC / 8; ++g) {
        const int cbase = g * 8;
        // stage depthwise weights for this channel group
        if (tid < 72)
            lwd[tid / 9][tid % 9] = w_dw[(cbase + tid / 9) * 9 + tid % 9];
        // stage 3 rows x 130 cols x 8 channels with zero padding
        for (int i = tid; i < 8 * 3 * 130; i += 128) {
            int ch = i / 390, rem = i % 390, r = rem / 130, c = rem % 130;
            int yy = y + r - 1, xx = x0 - 1 + c;
            __bf16 v = (__bf16)0.0f;
            if ((unsigned)yy < HH && (unsigned)xx < WW)
                v = qkv_bf[((size_t)(b * QKVC + cbase + ch) << 16) + yy * WW + xx];
            lh[ch][r][c] = v;
        }
        __syncthreads();

#pragma unroll
        for (int ch = 0; ch < 8; ++ch) {
            const int c = cbase + ch;
            float acc = 0.f;
#pragma unroll
            for (int r = 0; r < 3; ++r)
#pragma unroll
                for (int cc = 0; cc < 3; ++cc)
                    acc += lwd[ch][r * 3 + cc] * (float)lh[ch][r][tid + cc];
            if (c < 64) {
                lq[c][tid] = (__bf16)acc;
            } else if (c < 128) {
                lk[c - 64][tid] = (__bf16)acc;
            } else {
                int vc = c - 128;
                float il = illu[((size_t)(b * INNER + vc) << 16) + y * WW + x];
                vi_bf[((size_t)(b * INNER + vc) << 16) + y * WW + x] = (__bf16)(acc * il);
            }
        }
        __syncthreads();
    }

    // per-channel sum of squares (on the same bf16 values the Gram uses)
    {
        float s = 0.f;
        if (tid < 64) {
            for (int p = 0; p < 128; ++p) { float v = (float)lq[tid][p]; s += v * v; }
        } else {
            for (int p = 0; p < 128; ++p) { float v = (float)lk[tid - 64][p]; s += v * v; }
        }
        atomicAdd(&sumsq[b * 128 + tid], s);
    }

    // Gram via WMMA: wave wv handles head wv; 4 chunks of 32 pixels
    v8f acc = {};
#pragma unroll
    for (int c4 = 0; c4 < 4; ++c4) {
        int p0 = c4 * 32;
        v16bf a  = frag_a_16x32(&lq[0][0], 136, wv * 16, p0, lane);  // q[d][pix]
        v16bf bb = frag_b_colK(&lk[0][0], 136, p0, wv * 16, lane);   // k^T
        acc = wmma_bf16(a, bb, acc);
    }
    {
        int half = (lane >> 4) & 1;
        int e    = lane & 15;
#pragma unroll
        for (int r = 0; r < 8; ++r) {
            int d = r + 8 * half;
            atomicAdd(&G[((size_t)((b * HEADS + wv) * 16 + d) << 4) + e], acc[r]);
        }
    }
}

// =====================================================================
// K3: attn = softmax_e( G / (||q_d|| ||k_e||) * T ) ;  fold with w_out:
//     Cmat[b] = w_out(64x64) . blockdiag(attn[b,0..3])  -> bf16 [64][64]
// grid B, block 256
// =====================================================================
__global__ __launch_bounds__(256) void k3_softmax_fold(const float* __restrict__ G,
                                                       const float* __restrict__ sumsq,
                                                       const float* __restrict__ w_out,
                                                       const float* __restrict__ temperature,
                                                       __bf16* __restrict__ Cmat) {
    __shared__ float attn[HEADS][16][16];
    const int tid = threadIdx.x;
    const int b   = blockIdx.x;

    if (tid < 64) {
        int n = tid >> 4, d = tid & 15;
        float nq = fmaxf(sqrtf(sumsq[b * 128 + n * 16 + d]), 1e-12f);
        float tn = temperature[n];
        float row[16], mx = -1e30f;
#pragma unroll
        for (int e = 0; e < 16; ++e) {
            float nk = fmaxf(sqrtf(sumsq[b * 128 + 64 + n * 16 + e]), 1e-12f);
            float v  = G[((size_t)((b * HEADS + n) * 16 + d) << 4) + e] / (nq * nk) * tn;
            row[e] = v;
            mx = fmaxf(mx, v);
        }
        float s = 0.f;
#pragma unroll
        for (int e = 0; e < 16; ++e) { row[e] = expf(row[e] - mx); s += row[e]; }
        float inv = 1.f / s;
#pragma unroll
        for (int e = 0; e < 16; ++e) attn[n][d][e] = row[e] * inv;
    }
    __syncthreads();

    for (int i = tid; i < 64 * 64; i += 256) {
        int o = i >> 6, c = i & 63, n = c >> 4, e = c & 15;
        float s = 0.f;
#pragma unroll
        for (int d = 0; d < 16; ++d)
            s += w_out[o * 64 + n * 16 + d] * attn[n][d][e];
        Cmat[(size_t)b * 4096 + i] = (__bf16)s;
    }
}

// =====================================================================
// K4: out[b] = Cmat[b](64x64) @ vi[b](64 x HW)  -> f32 output
// grid (HW/256, B), block 256 (8 waves), 256-pixel tile per block.
// LDS staging via Tensor Data Mover (one issue per tile from wave 0;
// TDM pad fields reproduce the padded LDS row strides exactly).
// =====================================================================
__global__ __launch_bounds__(256) void k4_out_gemm(const __bf16* __restrict__ Cmat,
                                                   const __bf16* __restrict__ vi_bf,
                                                   float* __restrict__ out) {
    __shared__ __bf16 lc[INNER][72];     // A: [64][64]  rows 128B + 16B pad
    __shared__ __bf16 lvi[INNER][264];   // B: [64][256] rows 512B + 16B pad

    const int tid  = threadIdx.x;
    const int lane = tid & 31;
    const int wv   = tid >> 5;
    const int b    = blockIdx.y;
    const int p0   = blockIdx.x * 256;

#if HAVE_TDM
    if (wv == 0) {
        uint32_t lvi_off = (uint32_t)(uintptr_t)&lvi[0][0];
        uint32_t lc_off  = (uint32_t)(uintptr_t)&lc[0][0];
        const __bf16* vsrc = vi_bf + ((size_t)(b * INNER) << 16) + p0;
        const __bf16* csrc = Cmat + (size_t)b * 4096;
        // vi tile: tensor 65536x64, dim0_stride 65536, tile 256x64.
        // rows: 256 el = 128 DW -> pad_interval code 6 (2^7 DW); pad 4 DW -> ld 264.
        tdm_load_2d_bf16(lvi_off, vsrc, HW, 64, 256, 64, (uint64_t)HW, 6, 3);
        // Cmat tile: tensor 64x64, stride 64, tile 64x64.
        // rows: 64 el = 32 DW -> pad_interval code 4 (2^5 DW); pad 4 DW -> ld 72.
        tdm_load_2d_bf16(lc_off, csrc, 64, 64, 64, 64, 64ull, 4, 3);
        tdm_wait0();                      // TENSORcnt==0: both DMAs in LDS
    }
    __syncthreads();
#else
    for (int i = tid; i < 64 * 64; i += 256)
        lc[i >> 6][i & 63] = Cmat[(size_t)b * 4096 + i];
    for (int i = tid; i < INNER * 256; i += 256) {
        int c = i >> 8, p = i & 255;
        lvi[c][p] = vi_bf[((size_t)(b * INNER + c) << 16) + p0 + p];
    }
    __syncthreads();
#endif

    for (int t = 0; t < 8; ++t) {            // 64 tiles / 8 waves
        int task = wv * 8 + t;
        int m  = task & 3;                   // 4 M-tiles (64 out channels)
        int nn = task >> 2;                  // 16 N-subtiles (256 pixels)
        v8f acc = {};
#pragma unroll
        for (int k0 = 0; k0 < 64; k0 += 32) {
            v16bf a  = frag_a_16x32(&lc[0][0], 72, m * 16, k0, lane);
            v16bf bb = frag_b_rowK(&lvi[0][0], 264, k0, nn * 16, lane);
            acc = wmma_bf16(a, bb, acc);
        }
        int half = (lane >> 4) & 1;
        int pix  = p0 + nn * 16 + (lane & 15);
#pragma unroll
        for (int r = 0; r < 8; ++r) {
            int o = m * 16 + r + 8 * half;
            out[((size_t)(b * DIM + o) << 16) + pix] = acc[r];
        }
    }
}

// =====================================================================
// launch
// =====================================================================
extern "C" void kernel_launch(void* const* d_in, const int* in_sizes, int n_in,
                              void* d_out, int out_size, void* d_ws, size_t ws_size,
                              hipStream_t stream) {
    const float* x_in  = (const float*)d_in[0];
    const float* illu  = (const float*)d_in[1];
    const float* w_qkv = (const float*)d_in[2];
    const float* w_dw  = (const float*)d_in[3];
    const float* w_out = (const float*)d_in[4];
    const float* temp  = (const float*)d_in[5];
    float* out = (float*)d_out;

    // scratch layout
    char* ws = (char*)d_ws;
    const size_t qkv_bytes = (size_t)BB * QKVC * HW * sizeof(__bf16);   // 100.7 MB
    const size_t vi_bytes  = (size_t)BB * INNER * HW * sizeof(__bf16);  //  33.6 MB
    const size_t G_bytes   = (size_t)BB * HEADS * 16 * 16 * sizeof(float);
    const size_t ss_bytes  = (size_t)BB * 128 * sizeof(float);
    __bf16* qkv_bf = (__bf16*)ws;
    __bf16* vi_bf  = (__bf16*)(ws + qkv_bytes);
    float*  G      = (float*)(ws + qkv_bytes + vi_bytes);
    float*  sumsq  = (float*)(ws + qkv_bytes + vi_bytes + G_bytes);
    __bf16* Cmat   = (__bf16*)(ws + qkv_bytes + vi_bytes + G_bytes + ss_bytes);

    // zero reduction buffers (G + sumsq are contiguous)
    hipMemsetAsync(G, 0, G_bytes + ss_bytes, stream);

    k1_qkv_gemm<<<dim3(HW / 128, BB), 256, 0, stream>>>(x_in, w_qkv, qkv_bf);
    k2_dw_gram<<<dim3(WW / 128, HH, BB), 128, 0, stream>>>(qkv_bf, w_dw, illu, vi_bf, G, sumsq);
    k3_softmax_fold<<<dim3(BB), 256, 0, stream>>>(G, sumsq, w_out, temp, Cmat);
    k4_out_gemm<<<dim3(HW / 256, BB), 256, 0, stream>>>(Cmat, vi_bf, out);
}
